// LSTM_91122026152151
// MI455X (gfx1250) — compile-verified, matrix operand order checked by reading, not run on hardware
//
#include <hip/hip_runtime.h>

// LSTM forward for MI455X (gfx1250, wave32, WMMA bf16 16x16x32).
// Strategy:
//  - Pack x and weights ONCE into exact WMMA per-lane fragment layout (bf16).
//  - Gate-interleaved weight columns (col' = hid*4 + gate) so one 16x16 WMMA
//    output tile holds all 4 gates for 4 hidden units x 16 batch rows ->
//    fused GEMM + gate nonlinearities + state update in one kernel per step.
//  - 512 step launches provide inter-step synchronization (graph-friendly).

typedef __attribute__((ext_vector_type(16))) __bf16 v16bf;
typedef __attribute__((ext_vector_type(8)))  float  v8f;

#define B_  64
#define S_  512
#define I_  1024
#define H_  1024
#define NG  4096      // 4*H gate columns
#define NT  256       // NG/16 N-tiles
#define MT  4         // B/16  M-tiles
#define KT  32        // 1024/32 K-tiles per matrix

// K index inside a 32-wide K-tile for fragment element j in `lane`
// (CDNA5 ISA 7.12.2, 16-bit A 16x32 layout; B assumed mirrored with lane=N).
__device__ __forceinline__ int kmap(int lane, int j) {
    return (j & 7) + (((lane >> 4) & 1) << 3) + ((j >> 3) << 4);
}

__device__ __forceinline__ float lane_bcast(float v, int srcLane) {
    return __int_as_float(__builtin_amdgcn_ds_bpermute(srcLane << 2, __float_as_int(v)));
}

__device__ __forceinline__ float fsigmoid(float x) {
    return 1.0f / (1.0f + __expf(-x));
}
__device__ __forceinline__ float ftanh(float x) {
    float e = __expf(-2.0f * x);
    return (1.0f - e) / (1.0f + e);
}

// ---------------------------------------------------------------------------
// Pack weights into B-fragment layout, bf16, gate-interleaved columns.
// wx[k][n'] = W_gate[1024+k][hid],  wh[k][n'] = W_gate[k][hid],
//   n' = hid*4 + gate. Layout: [ktile][ntile][lane][16].
// 2^19 threads, 16 elements each.
__global__ __launch_bounds__(256)
void pack_weights(const float* __restrict__ Wi, const float* __restrict__ Wf,
                  const float* __restrict__ Wo, const float* __restrict__ Wc,
                  __bf16* __restrict__ wx, __bf16* __restrict__ wh) {
    int tid   = blockIdx.x * blockDim.x + threadIdx.x;
    int lane  = tid & 31;
    int ntile = (tid >> 5) & 255;
    int kt    = (tid >> 13) & 31;
    int mat   = (tid >> 18) & 1;          // 0 = Wx part, 1 = Wh part
    int nloc  = lane & 15;
    int gate  = nloc & 3;
    int hid   = ntile * 4 + (nloc >> 2);
    const float* src = (gate == 0) ? Wi : (gate == 1) ? Wf : (gate == 2) ? Wo : Wc;
    __bf16* dst = (mat ? wh : wx) + ((size_t)(kt * NT + ntile) * 32 + lane) * 16;
#pragma unroll
    for (int j = 0; j < 16; ++j) {
        int K   = kt * 32 + kmap(lane, j);
        int row = mat ? K : (H_ + K);     // Wh = rows [0,H), Wx = rows [H, I+H)
        dst[j] = (__bf16)src[(size_t)row * H_ + hid];
    }
}

__global__ __launch_bounds__(256)
void pack_bias(const float* __restrict__ bi, const float* __restrict__ bf,
               const float* __restrict__ bo, const float* __restrict__ bc,
               float* __restrict__ bias) {
    int n = blockIdx.x * blockDim.x + threadIdx.x;   // 0..4095
    int gate = n & 3, hid = n >> 2;
    const float* src = (gate == 0) ? bi : (gate == 1) ? bf : (gate == 2) ? bo : bc;
    bias[n] = src[hid];
}

// Pack x -> bf16 A-fragment layout: [t][mtile][ktile][lane][16].
// One thread per output element => fully coalesced writes.
__global__ __launch_bounds__(256)
void pack_x(const float* __restrict__ x, __bf16* __restrict__ xpack) {
    size_t tid = (size_t)blockIdx.x * blockDim.x + threadIdx.x;  // 2^25 total
    int j    = (int)(tid & 15);
    int lane = (int)((tid >> 4) & 31);
    int kt   = (int)((tid >> 9) & 31);
    int mt   = (int)((tid >> 14) & 3);
    int t    = (int)(tid >> 16);
    int b    = mt * 16 + (lane & 15);
    int K    = kt * 32 + kmap(lane, j);
    xpack[tid] = (__bf16)x[((size_t)b * S_ + t) * I_ + K];
}

// Pack h0 into A-fragment layout + copy c0.
__global__ __launch_bounds__(256)
void init_state(const float* __restrict__ h0, const float* __restrict__ c0,
                __bf16* __restrict__ hp0, float* __restrict__ cws) {
    int tid = blockIdx.x * blockDim.x + threadIdx.x;   // 131072 threads
    if (tid < 65536) {
        int j = tid & 15, lane = (tid >> 4) & 31, kt = (tid >> 9) & 31, mt = tid >> 14;
        int b = mt * 16 + (lane & 15);
        int K = kt * 32 + kmap(lane, j);
        hp0[tid] = (__bf16)h0[b * H_ + K];
    } else {
        int i = tid - 65536;
        cws[i] = c0[i];
    }
}

// ---------------------------------------------------------------------------
// One LSTM timestep, fully fused:
//   z = [x_t | h_{t-1}] @ [Wx; Wh] + b   (bf16 WMMA, f32 accum)
//   gates -> c_t, h_t;  h_t written to output + re-packed for next step.
// 1024 waves: wave = mtile*256 + ntile. 16x16 tile = 16 batch rows x
// (4 hidden units x 4 gates) thanks to gate-interleaved columns.
__global__ __launch_bounds__(256)
void lstm_step(const __bf16* __restrict__ xpack,
               const __bf16* __restrict__ wx,
               const __bf16* __restrict__ wh,
               const float*  __restrict__ bias,
               const __bf16* __restrict__ hin,
               __bf16*       __restrict__ hout,
               float*        __restrict__ cws,
               float*        __restrict__ out,
               int t) {
    int lane  = threadIdx.x & 31;
    int wave  = (blockIdx.x * blockDim.x + threadIdx.x) >> 5;  // 0..1023
    int ntile = wave & (NT - 1);
    int mtile = wave >> 8;

    v8f acc = {0.f, 0.f, 0.f, 0.f, 0.f, 0.f, 0.f, 0.f};

    // --- x_t @ Wx ---
    const v16bf* AX = ((const v16bf*)xpack) + (((size_t)t * MT + mtile) * KT) * 32 + lane;
    const v16bf* BX = ((const v16bf*)wx) + (size_t)ntile * 32 + lane;
#pragma unroll 4
    for (int kt = 0; kt < KT; ++kt) {
        v16bf a = AX[(size_t)kt * 32];
        v16bf b = BX[(size_t)kt * NT * 32];
        acc = __builtin_amdgcn_wmma_f32_16x16x32_bf16(false, a, false, b,
                                                      (short)0, acc, false, false);
    }
    // --- h_{t-1} @ Wh ---
    const v16bf* AH = ((const v16bf*)hin) + ((size_t)mtile * KT) * 32 + lane;
    const v16bf* BH = ((const v16bf*)wh) + (size_t)ntile * 32 + lane;
#pragma unroll 4
    for (int kt = 0; kt < KT; ++kt) {
        v16bf a = AH[(size_t)kt * 32];
        v16bf b = BH[(size_t)kt * NT * 32];
        acc = __builtin_amdgcn_wmma_f32_16x16x32_bf16(false, a, false, b,
                                                      (short)0, acc, false, false);
    }

    // C/D layout: VGPR r, lanes 0-15: M=r,N=lane ; lanes 16-31: M=r+8,N=lane-16
    int nloc  = lane & 15;
    float bn  = bias[ntile * 16 + nloc];
    int gate  = nloc & 3;
    int hid   = ntile * 4 + (nloc >> 2);
    int mbase = (lane >> 4) << 3;
    int grp   = lane & ~3;                 // lane holding gate 0 of this group

    (void)gate;
#pragma unroll
    for (int r = 0; r < 8; ++r) {
        float z  = acc[r] + bn;
        float zi = lane_bcast(z, grp + 0);
        float zf = lane_bcast(z, grp + 1);
        float zo = lane_bcast(z, grp + 2);
        float zg = lane_bcast(z, grp + 3);
        float it = fsigmoid(zi);
        float ft = fsigmoid(zf);
        float ot = fsigmoid(zo);
        float gt = ftanh(zg);

        int m = mbase + r;                 // 0..15
        int b = mtile * 16 + m;
        float cold = cws[b * H_ + hid];
        float cnew = ft * cold + it * gt;
        float hnew = ot * ftanh(cnew);

        if ((lane & 3) == 0) {             // one lane per (b, hid) commits
            cws[b * H_ + hid] = cnew;
            out[((size_t)b * S_ + t) * H_ + hid] = hnew;
            // re-pack h into A-fragment layout for next step
            int kk = hid & 31, kt2 = hid >> 5;
            int laneW = m + (((kk >> 3) & 1) << 4);
            int jW = (kk & 7) + ((kk >> 4) << 3);
            hout[(((size_t)mtile * KT + kt2) * 32 + laneW) * 16 + jW] = (__bf16)hnew;
            if (t == S_ - 1) {
                out[(size_t)B_ * S_ * H_ + b * H_ + hid] = hnew;                 // h_T
                out[(size_t)B_ * S_ * H_ + (size_t)B_ * H_ + b * H_ + hid] = cnew; // c_T
            }
        }
    }
}

// ---------------------------------------------------------------------------
extern "C" void kernel_launch(void* const* d_in, const int* in_sizes, int n_in,
                              void* d_out, int out_size, void* d_ws, size_t ws_size,
                              hipStream_t stream) {
    const float* x  = (const float*)d_in[0];
    const float* h0 = (const float*)d_in[1];
    const float* c0 = (const float*)d_in[2];
    const float* Wi = (const float*)d_in[3];
    const float* bi = (const float*)d_in[4];
    const float* Wf = (const float*)d_in[5];
    const float* bf = (const float*)d_in[6];
    const float* Wo = (const float*)d_in[7];
    const float* bo = (const float*)d_in[8];
    const float* Wc = (const float*)d_in[9];
    const float* bc = (const float*)d_in[10];
    float* out = (float*)d_out;

    char* ws = (char*)d_ws;
    const size_t XPACK_B = (size_t)B_ * S_ * I_ * 2;      // 67,108,864
    const size_t W_B     = (size_t)I_ * NG * 2;           //  8,388,608 each
    __bf16* xpack = (__bf16*)(ws);
    __bf16* wx    = (__bf16*)(ws + XPACK_B);
    __bf16* wh    = (__bf16*)(ws + XPACK_B + W_B);
    float*  bias  = (float*)(ws + XPACK_B + 2 * W_B);
    __bf16* hp0   = (__bf16*)(ws + XPACK_B + 2 * W_B + NG * 4);
    __bf16* hp1   = hp0 + (size_t)B_ * H_;
    float*  cws   = (float*)((char*)hp1 + (size_t)B_ * H_ * 2);

    pack_weights<<<(2 * KT * NT * 32) / 256, 256, 0, stream>>>(Wi, Wf, Wo, Wc, wx, wh);
    pack_bias<<<NG / 256, 256, 0, stream>>>(bi, bf, bo, bc, bias);
    pack_x<<<((size_t)B_ * S_ * I_) / 256, 256, 0, stream>>>(x, xpack);
    init_state<<<(2 * B_ * H_) / 256, 256, 0, stream>>>(h0, c0, hp0, cws);

    for (int t = 0; t < S_; ++t) {
        const __bf16* hin = (t & 1) ? hp1 : hp0;
        __bf16* hout      = (t & 1) ? hp0 : hp1;
        lstm_step<<<(MT * NT * 32) / 256, 256, 0, stream>>>(
            xpack, wx, wh, bias, hin, hout, cws, out, t);
    }
}
